// TransformerBlock_85418309583081
// MI455X (gfx1250) — matrix operand order repaired and checked by reading, not compiled, not run on hardware
//
#include <hip/hip_runtime.h>
#include <math.h>

// ---------------------------------------------------------------------------
// MI455X (gfx1250) differential-attention transformer block.
// All matmuls (qkv, QK^T, PV, w_out, FFN1, FFN2) run on
// v_wmma_f32_16x16x32_bf16 (bf16 in, f32 accumulate). Softmax/RoPE/norms in
// f32. Fragment layouts per CDNA5 ISA:
//   16-bit A: lane-half K-base {0,8}; elems 0..7 -> K base..base+7,
//             elems 8..15 -> K 16+base..16+base+7; row = lane&15.
//   16-bit B: same K order as A (lane-half K-base {0,8}); N = lane&15
//             (i.e. B fragment = 8 bf16 at k=base + 8 at k=16+base of an [n][k] row).
//   f32 C/D : VGPR v -> row v + 8*(lane>=16), col = lane&15.
// Mainloops are LDS double-buffered. Contiguous tile copies use the CDNA5
// async data mover path (global_load_async_to_lds_b128 + s_wait_asynccnt);
// transposed copies use packed b32 DS stores.
// ---------------------------------------------------------------------------

typedef __attribute__((ext_vector_type(16))) __bf16 v16bf;
typedef __attribute__((ext_vector_type(8)))  float  v8f;

struct BF16x16 { uint4 a; uint4 b; };   // 32 bytes == one v16bf fragment

#define Bdim 2
#define Sdim 2048
#define Ddim 1024
#define Hdim 8

// --- CDNA5 async global->LDS copy (CDNA5 ISA) --------
// Per-lane: LDS[lds_off] = MEM[gptr], 16 bytes, tracked by ASYNCcnt.
__device__ __forceinline__ void async_load_b128(unsigned lds_off, const void* gptr) {
  asm volatile("global_load_async_to_lds_b128 %0, %1, off"
               :: "v"(lds_off), "v"((unsigned long long)(size_t)gptr)
               : "memory");
}
__device__ __forceinline__ void wait_async() {
  asm volatile("s_wait_asynccnt 0x0" ::: "memory");
}
__device__ __forceinline__ unsigned lds_off_of(const void* p) {
  // Generic pointers into the LDS aperture carry the LDS offset in addr[31:0].
  return (unsigned)(size_t)p;
}

// ------------------------------ f32 -> bf16 --------------------------------
__global__ void f32_to_bf16(const float* __restrict__ in, __bf16* __restrict__ out, size_t n) {
  size_t i = (size_t)blockIdx.x * blockDim.x + threadIdx.x;
  size_t stride = (size_t)gridDim.x * blockDim.x;
  for (; i < n; i += stride) out[i] = (__bf16)in[i];
}

// ------------------------------ lambda scalar ------------------------------
__global__ void compute_lambda(const float* __restrict__ lq1, const float* __restrict__ lk1,
                               const float* __restrict__ lq2, const float* __restrict__ lk2,
                               float* __restrict__ lam) {
  int lane = threadIdx.x;                       // 32 threads, HD=64
  float s1 = lq1[lane] * lk1[lane] + lq1[lane + 32] * lk1[lane + 32];
  float s2 = lq2[lane] * lk2[lane] + lq2[lane + 32] * lk2[lane + 32];
  for (int off = 1; off < 32; off <<= 1) {
    s1 += __shfl_xor(s1, off, 32);
    s2 += __shfl_xor(s2, off, 32);
  }
  if (lane == 0) {
    const float LI = 0.8f - 0.6f * __expf(-0.6f);   // LAMBDA_INIT, DEPTH=2
    lam[0] = __expf(s1) - __expf(s2) + LI;
  }
}

// ------------------------------ bf16 GEMM ----------------------------------
// C[M,N] = act(A[M,K] x B[K,N] + bias). Block tile 64x256, 8 waves (2x4),
// each wave 32x64 via 2x4 WMMA accumulators. K-step 32, double-buffered LDS.
// A tile copied via async DMA; B tile transposed via packed b32 DS stores.
#define GM_BM 64
#define GM_BN 256
#define GM_BK 32

template <int ACT, bool HASBIAS, bool OUTF, bool OUTB>
__global__ __launch_bounds__(256) void gemm_bf16(
    const __bf16* __restrict__ A, const __bf16* __restrict__ B,
    const float* __restrict__ bias, float* __restrict__ Cf,
    __bf16* __restrict__ Cb, int M, int N, int K) {
  __shared__ __bf16 As[2][GM_BM * GM_BK];     // [m][k], 2 x 4 KB
  __shared__ __bf16 Bs[2][GM_BN * GM_BK];     // [n][k] (transposed), 2 x 16 KB

  const int t = threadIdx.x;
  const int wave = t >> 5, lane = t & 31;
  const int wm = wave >> 2, wn = wave & 3;
  const int half = lane >> 4, l16 = lane & 15;
  const int m0 = blockIdx.y * GM_BM;
  const int n0 = blockIdx.x * GM_BN;
  const int nk = K / GM_BK;

  const v8f zero = {};
  v8f acc[2][4];
  for (int i = 0; i < 2; ++i)
    for (int j = 0; j < 4; ++j) acc[i][j] = zero;

  auto load_tile = [&](int kt, int buf) {
    {   // A tile 64x32: one async b128 per lane (CDNA5 async data mover).
      int r = t >> 2, cseg = (t & 3) * 8;
      async_load_b128(lds_off_of(&As[buf][r * GM_BK + cseg]),
                      A + (size_t)(m0 + r) * K + kt + cseg);
    }
#pragma unroll
    for (int pass = 0; pass < 2; ++pass) {   // B tile 32x256 -> Bs[n][k]
      int idx = t + pass * 256;
      int pr = idx >> 5;                      // k-row pair 0..15
      int c = (idx & 31) * 8;                 // n col
      const __bf16* src = &B[(size_t)(kt + pr * 2) * N + n0 + c];
      uint4 u0 = *(const uint4*)src;
      uint4 u1 = *(const uint4*)(src + N);
      if (kt + GM_BK < K)
        __builtin_prefetch(src + (size_t)GM_BK * N, 0, 1);  // global_prefetch
      const unsigned short* e0 = (const unsigned short*)&u0;
      const unsigned short* e1 = (const unsigned short*)&u1;
#pragma unroll
      for (int j = 0; j < 8; ++j) {
        unsigned int w = (unsigned int)e0[j] | ((unsigned int)e1[j] << 16);
        *(unsigned int*)&Bs[buf][(c + j) * GM_BK + pr * 2] = w;
      }
    }
  };

  load_tile(0, 0);
  wait_async();
  __syncthreads();

  for (int kti = 0; kti < nk; ++kti) {
    const int cur = kti & 1;
    if (kti + 1 < nk) load_tile((kti + 1) * GM_BK, cur ^ 1);

    v16bf af[2], bfv[4];
#pragma unroll
    for (int i = 0; i < 2; ++i) {
      const __bf16* p = &As[cur][(wm * 32 + i * 16 + l16) * GM_BK] + half * 8;
      BF16x16 tmp;
      tmp.a = *(const uint4*)p;
      tmp.b = *(const uint4*)(p + 16);
      af[i] = __builtin_bit_cast(v16bf, tmp);
    }
#pragma unroll
    for (int j = 0; j < 4; ++j) {
      const __bf16* p = &Bs[cur][(wn * 64 + j * 16 + l16) * GM_BK] + half * 8;
      BF16x16 tmp;
      tmp.a = *(const uint4*)p;
      tmp.b = *(const uint4*)(p + 16);
      bfv[j] = __builtin_bit_cast(v16bf, tmp);
    }
#pragma unroll
    for (int i = 0; i < 2; ++i)
#pragma unroll
      for (int j = 0; j < 4; ++j)
        acc[i][j] = __builtin_amdgcn_wmma_f32_16x16x32_bf16(
            false, af[i], false, bfv[j], (short)0, acc[i][j], false, false);

    wait_async();          // next tile's DMA has overlapped with the WMMAs
    __syncthreads();
  }

  // Epilogue. C layout: VGPR v -> row v + 8*half, col = lane&15.
#pragma unroll
  for (int i = 0; i < 2; ++i) {
#pragma unroll
    for (int j = 0; j < 4; ++j) {
#pragma unroll
      for (int v = 0; v < 8; ++v) {
        int row = m0 + wm * 32 + i * 16 + v + half * 8;
        int col = n0 + wn * 64 + j * 16 + l16;
        float c = acc[i][j][v];
        if (HASBIAS) c += bias[col];
        if (ACT == 1) c = c * __builtin_amdgcn_rcpf(1.0f + __expf(-c));  // SiLU
        size_t idx = (size_t)row * N + col;
        if (OUTF) Cf[idx] = c;
        if (OUTB) Cb[idx] = (__bf16)c;
      }
    }
  }
}

// --------------------------- RoPE + split + norms --------------------------
// qkv f32 [B*S, 3D] -> Q1/K1 bf16 [B,H,S,64], Q2n/K2n = rmsnorm(rope+sig)
// bf16 [B,H,S,64], V bf16 [B,H,S,128]. Q streams are pre-scaled by 1/sqrt(HD)
// so the attention kernel applies no score scaling. One block per token.
__global__ __launch_bounds__(256) void rope_split(
    const float* __restrict__ qkv, const float* __restrict__ sigmas,
    const float* __restrict__ lnq_w, const float* __restrict__ lnq_b,
    __bf16* __restrict__ Q1, __bf16* __restrict__ K1,
    __bf16* __restrict__ Q2, __bf16* __restrict__ K2,
    __bf16* __restrict__ Vt) {
  const int bs = blockIdx.x;
  const int b = bs / Sdim, s = bs % Sdim;
  const int wave = threadIdx.x >> 5, lane = threadIdx.x & 31;
  const float* row = qkv + (size_t)bs * (3 * Ddim);
  const int d0 = 2 * lane;                 // lanes 0-15 -> rot dims, 16-31 pass

  float c = 1.0f, sn = 0.0f;
  if (lane < 16) {
    float th = (float)s * __powf(10000.0f, -(float)lane / 16.0f);
    c = __cosf(th);
    sn = __sinf(th);
  }

#pragma unroll
  for (int which = 0; which < 2; ++which) {        // 0=q, 1=k
    const float* base = row + which * Ddim;
    const float qscale = which ? 1.0f : 0.125f;    // fold 1/sqrt(64) into Q
#pragma unroll
    for (int jj = 0; jj < 2; ++jj) {               // sub-head j in (H,2,HD)
      int hh = wave * 2 + jj;
      float x0 = base[hh * 64 + d0];
      float x1 = base[hh * 64 + d0 + 1];
      float y0, y1;
      if (lane < 16) { y0 = x0 * c - x1 * sn; y1 = x1 * c + x0 * sn; }
      else           { y0 = x0;               y1 = x1; }
      size_t o = ((size_t)(b * Hdim + wave) * Sdim + s) * 64;
      if (jj == 0) {
        __bf16* dst = (which ? K1 : Q1) + o;
        dst[d0] = (__bf16)(y0 * qscale);
        dst[d0 + 1] = (__bf16)(y1 * qscale);
      } else {
        y0 += sigmas[b * 64 + d0];
        y1 += sigmas[b * 64 + d0 + 1];
        float ss = y0 * y0 + y1 * y1;
        for (int off = 1; off < 32; off <<= 1) ss += __shfl_xor(ss, off, 32);
        float sc = rsqrtf(ss * (1.0f / 64.0f) + 1e-8f);
        float z0 = y0 * sc * lnq_w[d0] + lnq_b[d0];
        float z1 = y1 * sc * lnq_w[d0 + 1] + lnq_b[d0 + 1];
        __bf16* dst = (which ? K2 : Q2) + o;
        dst[d0] = (__bf16)(z0 * qscale);
        dst[d0 + 1] = (__bf16)(z1 * qscale);
      }
    }
  }
  // V copy: wave w handles head w (128 dims).
#pragma unroll
  for (int r = 0; r < 4; ++r) {
    int d = r * 32 + lane;
    Vt[((size_t)(b * Hdim + wave) * Sdim + s) * 128 + d] =
        (__bf16)row[2 * Ddim + wave * 128 + d];
  }
}

// ------------------------------ flash attention ----------------------------
// One block = 64 q rows of one (b,h,stream). 4 waves x 16 rows. Key tiles of
// 32 with online softmax, double-buffered K/V tiles (K via async DMA).
// Causal masking only in the <=2 diagonal key blocks (wave-uniform branch).
__global__ __launch_bounds__(128) void flash_attn(
    const __bf16* __restrict__ Qa, const __bf16* __restrict__ Ka,
    const __bf16* __restrict__ Qb, const __bf16* __restrict__ Kb,
    const __bf16* __restrict__ Vt, float* __restrict__ O1,
    float* __restrict__ O2) {
  __shared__ __bf16 Ks[2][32 * 64];      // [key][dim] == B-frag layout for QK^T
  __shared__ __bf16 Vs[2][128 * 32];     // [dim][key] == B-frag layout for PV
  __shared__ __bf16 Ps[4 * 16 * 32];     // per-wave P scratch

  const int t = threadIdx.x, wave = t >> 5, lane = t & 31;
  const int half = lane >> 4, l16 = lane & 15;
  const int q0 = blockIdx.x * 64;
  const int bh = blockIdx.y;                       // b*H + h
  const int strm = blockIdx.z;

  const __bf16* Q = strm ? Qb : Qa;
  const __bf16* Kg = strm ? Kb : Ka;
  float* O = strm ? O2 : O1;
  const __bf16* Qrow = Q + (size_t)bh * Sdim * 64;
  const __bf16* Krow = Kg + (size_t)bh * Sdim * 64;
  const __bf16* Vrow = Vt + (size_t)bh * Sdim * 128;

  // Q fragments (A-matrix, 16x32 per K-chunk) live in registers all loop long.
  v16bf qf[2];
  {
    int row = q0 + wave * 16 + l16;
#pragma unroll
    for (int cchunk = 0; cchunk < 2; ++cchunk) {
      const __bf16* p = Qrow + (size_t)row * 64 + cchunk * 32 + half * 8;
      BF16x16 tmp;
      tmp.a = *(const uint4*)p;
      tmp.b = *(const uint4*)(p + 16);
      qf[cchunk] = __builtin_bit_cast(v16bf, tmp);
    }
  }

  const v8f zero = {};
  v8f o[8];
  float m[8], l[8];
#pragma unroll
  for (int nt = 0; nt < 8; ++nt) o[nt] = zero;
#pragma unroll
  for (int v = 0; v < 8; ++v) { m[v] = -1e30f; l[v] = 0.0f; }

  // Cooperative K/V tile load (128 threads). K tile via async DMA (natural
  // layout); V transposed into [dim][key] with packed b32 DS stores.
  auto load_kv = [&](int kb, int buf) {
    const int k0 = kb * 32;
    {
      int key = t >> 2, seg = t & 3;
      const __bf16* src = Krow + (size_t)(k0 + key) * 64 + seg * 16;
      unsigned dst = lds_off_of(Ks[buf] + key * 64 + seg * 16);
      async_load_b128(dst, src);
      async_load_b128(dst + 16, src + 8);
    }
    {
      int kp = t >> 3, seg = t & 7;           // key pair, 16-dim segment
      const __bf16* s0 = Vrow + (size_t)(k0 + kp * 2) * 128 + seg * 16;
      uint4 kv0[2], kv1[2];
      kv0[0] = *(const uint4*)s0;
      kv0[1] = *(const uint4*)(s0 + 8);
      kv1[0] = *(const uint4*)(s0 + 128);
      kv1[1] = *(const uint4*)(s0 + 136);
      const unsigned short* e0 = (const unsigned short*)kv0;
      const unsigned short* e1 = (const unsigned short*)kv1;
#pragma unroll
      for (int j = 0; j < 16; ++j) {
        unsigned int w = (unsigned int)e0[j] | ((unsigned int)e1[j] << 16);
        *(unsigned int*)&Vs[buf][(seg * 16 + j) * 32 + kp * 2] = w;
      }
    }
  };

  const int nkb = q0 / 32 + 2;                    // causal key-block bound
  load_kv(0, 0);
  wait_async();
  __syncthreads();

  for (int kb = 0; kb < nkb; ++kb) {
    const int cur = kb & 1;
    const int k0 = kb * 32;
    if (kb + 1 < nkb) load_kv(kb + 1, cur ^ 1);

    // Scores: 16 q x 32 keys = 2 N-tiles x 2 K-chunks (Q pre-scaled).
    v8f s[2];
    s[0] = zero;
    s[1] = zero;
#pragma unroll
    for (int nc = 0; nc < 2; ++nc) {
#pragma unroll
      for (int cchunk = 0; cchunk < 2; ++cchunk) {
        const __bf16* p = Ks[cur] + (nc * 16 + l16) * 64 + cchunk * 32 + half * 8;
        BF16x16 tmp;
        tmp.a = *(const uint4*)p;
        tmp.b = *(const uint4*)(p + 16);
        v16bf kf = __builtin_bit_cast(v16bf, tmp);
        s[nc] = __builtin_amdgcn_wmma_f32_16x16x32_bf16(
            false, qf[cchunk], false, kf, (short)0, s[nc], false, false);
      }
    }

    // Causal mask only in diagonal blocks (wave-uniform condition).
    const int qbase = q0 + wave * 16 + half * 8;
    if (k0 + 31 > q0 + wave * 16) {
#pragma unroll
      for (int v = 0; v < 8; ++v) {
        int qi = qbase + v;
#pragma unroll
        for (int nc = 0; nc < 2; ++nc) {
          int ki = k0 + nc * 16 + l16;
          s[nc][v] = (ki <= qi) ? s[nc][v] : -1e30f;
        }
      }
    }

    // Online softmax; rows live across 16-lane halves.
#pragma unroll
    for (int v = 0; v < 8; ++v) {
      float rmax = fmaxf(s[0][v], s[1][v]);
      for (int off = 1; off < 16; off <<= 1) rmax = fmaxf(rmax, __shfl_xor(rmax, off, 32));
      float mnew = fmaxf(m[v], rmax);
      float p0 = __expf(s[0][v] - mnew);
      float p1 = __expf(s[1][v] - mnew);
      float rs = p0 + p1;
      for (int off = 1; off < 16; off <<= 1) rs += __shfl_xor(rs, off, 32);
      float alpha = __expf(m[v] - mnew);
      l[v] = l[v] * alpha + rs;
      m[v] = mnew;
      s[0][v] = p0;
      s[1][v] = p1;
#pragma unroll
      for (int nt = 0; nt < 8; ++nt) o[nt][v] *= alpha;
    }

    // P (C layout) -> per-wave LDS -> A-fragment layout.
    __bf16* Pw = Ps + wave * 512;
#pragma unroll
    for (int v = 0; v < 8; ++v) {
      Pw[(v + half * 8) * 32 + l16] = (__bf16)s[0][v];
      Pw[(v + half * 8) * 32 + 16 + l16] = (__bf16)s[1][v];
    }
    v16bf pf;
    {
      const __bf16* p = Pw + l16 * 32 + half * 8;
      BF16x16 tmp;
      tmp.a = *(const uint4*)p;
      tmp.b = *(const uint4*)(p + 16);
      pf = __builtin_bit_cast(v16bf, tmp);
    }
    // PV: 8 N-tiles over the 128-wide value dim, K = 32 keys per WMMA.
#pragma unroll
    for (int nt = 0; nt < 8; ++nt) {
      const __bf16* p = Vs[cur] + (nt * 16 + l16) * 32 + half * 8;
      BF16x16 tmp;
      tmp.a = *(const uint4*)p;
      tmp.b = *(const uint4*)(p + 16);
      v16bf vf = __builtin_bit_cast(v16bf, tmp);
      o[nt] = __builtin_amdgcn_wmma_f32_16x16x32_bf16(
          false, pf, false, vf, (short)0, o[nt], false, false);
    }

    wait_async();          // next K tile's DMA overlapped with compute
    __syncthreads();
  }

  // Normalize and write O [B,S,H,128] f32.
  const int b = bh >> 3, h = bh & 7;
#pragma unroll
  for (int v = 0; v < 8; ++v) {
    int qi = q0 + wave * 16 + half * 8 + v;
    float inv = __builtin_amdgcn_rcpf(l[v]);
    float* orow = O + ((size_t)(b * Sdim + qi) * Hdim + h) * 128;
#pragma unroll
    for (int nt = 0; nt < 8; ++nt) orow[nt * 16 + l16] = o[nt][v] * inv;
  }
}

// -------------------- diff combine: rmsnorm(a1 - lam*a2) -------------------
__global__ __launch_bounds__(256) void diff_combine(
    const float* __restrict__ A1, const float* __restrict__ A2,
    const float* __restrict__ lamp, const float* __restrict__ w,
    const float* __restrict__ bvec, float* __restrict__ outf,
    __bf16* __restrict__ outb) {
  const int wave = threadIdx.x >> 5, lane = threadIdx.x & 31;
  const size_t g = (size_t)blockIdx.x * 8 + wave;     // (b*S+s)*H + h
  const float lam = lamp[0];
  const float LI = 0.8f - 0.6f * __expf(-0.6f);
  float y[4];
  float ss = 0.0f;
#pragma unroll
  for (int i = 0; i < 4; ++i) {
    int d = lane + i * 32;
    size_t idx = g * 128 + d;
    y[i] = A1[idx] - lam * A2[idx];
    ss += y[i] * y[i];
  }
  for (int off = 1; off < 32; off <<= 1) ss += __shfl_xor(ss, off, 32);
  float sc = rsqrtf(ss * (1.0f / 128.0f) + 1e-8f);
#pragma unroll
  for (int i = 0; i < 4; ++i) {
    int d = lane + i * 32;
    size_t idx = g * 128 + d;
    float v = (y[i] * sc * w[d] + bvec[d]) * (1.0f - LI);
    outf[idx] = v;
    outb[idx] = (__bf16)v;
  }
}

// ----------------------- residual + layernorm (D=1024) ---------------------
__global__ __launch_bounds__(256) void residual_layernorm(
    const float* __restrict__ A, const float* __restrict__ R,
    const float* __restrict__ w, const float* __restrict__ bvec,
    float* __restrict__ outf, __bf16* __restrict__ outb) {
  __shared__ float red[16];
  const int row = blockIdx.x;
  const float* a = A + (size_t)row * Ddim;
  const float* r = R + (size_t)row * Ddim;
  float z[4];
  float s = 0.0f, s2 = 0.0f;
#pragma unroll
  for (int i = 0; i < 4; ++i) {
    int idx = threadIdx.x + i * 256;
    z[i] = a[idx] + r[idx];
    s += z[i];
    s2 += z[i] * z[i];
  }
  for (int off = 1; off < 32; off <<= 1) {
    s += __shfl_xor(s, off, 32);
    s2 += __shfl_xor(s2, off, 32);
  }
  const int wave = threadIdx.x >> 5, lane = threadIdx.x & 31;
  if (lane == 0) { red[wave] = s; red[8 + wave] = s2; }
  __syncthreads();
  if (threadIdx.x == 0) {
    float ts = 0.0f, ts2 = 0.0f;
    for (int i = 0; i < 8; ++i) { ts += red[i]; ts2 += red[8 + i]; }
    red[0] = ts;
    red[8] = ts2;
  }
  __syncthreads();
  float mean = red[0] * (1.0f / Ddim);
  float var = red[8] * (1.0f / Ddim) - mean * mean;
  float inv = rsqrtf(var + 1e-5f);
#pragma unroll
  for (int i = 0; i < 4; ++i) {
    int idx = threadIdx.x + i * 256;
    float y = (z[i] - mean) * inv * w[idx] + bvec[idx];
    if (outf) outf[(size_t)row * Ddim + idx] = y;
    if (outb) outb[(size_t)row * Ddim + idx] = (__bf16)y;
  }
}

// ------------------------------- launcher ----------------------------------
extern "C" void kernel_launch(void* const* d_in, const int* in_sizes, int n_in,
                              void* d_out, int out_size, void* d_ws, size_t ws_size,
                              hipStream_t stream) {
  (void)in_sizes; (void)n_in; (void)out_size; (void)ws_size;
  const float* x      = (const float*)d_in[0];
  const float* sigmas = (const float*)d_in[1];
  const float* w_qkv  = (const float*)d_in[2];
  const float* w_out  = (const float*)d_in[3];
  const float* lq1    = (const float*)d_in[4];
  const float* lk1    = (const float*)d_in[5];
  const float* lq2    = (const float*)d_in[6];
  const float* lk2    = (const float*)d_in[7];
  const float* ln_w   = (const float*)d_in[8];
  const float* ln_b   = (const float*)d_in[9];
  const float* lnq_w  = (const float*)d_in[10];
  const float* lnq_b  = (const float*)d_in[11];
  const float* ln1_w  = (const float*)d_in[12];
  const float* ln1_b  = (const float*)d_in[13];
  const float* ln2_w  = (const float*)d_in[14];
  const float* ln2_b  = (const float*)d_in[15];
  const float* w1     = (const float*)d_in[16];
  const float* b1     = (const float*)d_in[17];
  const float* w2     = (const float*)d_in[18];
  const float* b2     = (const float*)d_in[19];
  float* out = (float*)d_out;

  const size_t M = (size_t)Bdim * Sdim;          // 4096 rows
  char* ws = (char*)d_ws;
  size_t off = 0;
  auto alloc = [&](size_t bytes) -> char* {
    char* p = ws + off;
    off += (bytes + 255) & ~(size_t)255;
    return p;
  };

  __bf16* wqkv_b = (__bf16*)alloc((size_t)Ddim * 3 * Ddim * 2);
  __bf16* wout_b = (__bf16*)alloc((size_t)Ddim * Ddim * 2);
  __bf16* w1_b   = (__bf16*)alloc((size_t)Ddim * 4 * Ddim * 2);
  __bf16* w2_b   = (__bf16*)alloc((size_t)4 * Ddim * Ddim * 2);
  __bf16* act_b  = (__bf16*)alloc(M * Ddim * 2);           // xbf -> a_bf -> h_bf
  char*   big    = alloc(M * 3 * Ddim * 4);                // qkv f32 -> attn1/2 -> ffn1 bf16
  __bf16* Q1 = (__bf16*)alloc((size_t)Bdim * Hdim * Sdim * 64 * 2);
  __bf16* K1 = (__bf16*)alloc((size_t)Bdim * Hdim * Sdim * 64 * 2);
  __bf16* Q2 = (__bf16*)alloc((size_t)Bdim * Hdim * Sdim * 64 * 2);
  __bf16* K2 = (__bf16*)alloc((size_t)Bdim * Hdim * Sdim * 64 * 2);
  __bf16* Vt = (__bf16*)alloc((size_t)Bdim * Hdim * Sdim * 128 * 2);
  float*  f32a = (float*)alloc(M * Ddim * 4);              // "a" -> h
  float*  hbuf = (float*)alloc(M * Ddim * 4);              // a_proj -> ffn2
  float*  lam  = (float*)alloc(256);

  float*  qkv_f  = (float*)big;
  float*  attn1  = (float*)big;
  float*  attn2  = (float*)(big + M * Ddim * 4);
  __bf16* ffn1_b = (__bf16*)big;

  // 1) bf16 conversions (weights + x) and lambda scalar.
  f32_to_bf16<<<1024, 256, 0, stream>>>(x, act_b, M * Ddim);
  f32_to_bf16<<<1024, 256, 0, stream>>>(w_qkv, wqkv_b, (size_t)Ddim * 3 * Ddim);
  f32_to_bf16<<<1024, 256, 0, stream>>>(w_out, wout_b, (size_t)Ddim * Ddim);
  f32_to_bf16<<<1024, 256, 0, stream>>>(w1, w1_b, (size_t)Ddim * 4 * Ddim);
  f32_to_bf16<<<1024, 256, 0, stream>>>(w2, w2_b, (size_t)4 * Ddim * Ddim);
  compute_lambda<<<1, 32, 0, stream>>>(lq1, lk1, lq2, lk2, lam);

  // 2) qkv = x @ w_qkv   (4096 x 3072 x 1024)
  gemm_bf16<0, false, true, false><<<dim3(3 * Ddim / GM_BN, M / GM_BM), 256, 0, stream>>>(
      act_b, wqkv_b, nullptr, qkv_f, nullptr, (int)M, 3 * Ddim, Ddim);

  // 3) RoPE + split + q2/k2 rmsnorm + v pack (Q pre-scaled by 1/8).
  rope_split<<<(int)M, 256, 0, stream>>>(qkv_f, sigmas, lnq_w, lnq_b,
                                         Q1, K1, Q2, K2, Vt);

  // 4) Both differential-attention streams (grid.z = stream).
  flash_attn<<<dim3(Sdim / 64, Bdim * Hdim, 2), 128, 0, stream>>>(
      Q1, K1, Q2, K2, Vt, attn1, attn2);

  // 5) a = rmsnorm(attn1 - lam*attn2) * (1-LI), f32 + bf16.
  diff_combine<<<(int)(M * Hdim / 8), 256, 0, stream>>>(
      attn1, attn2, lam, ln_w, ln_b, f32a, act_b);

  // 6) a_proj = a @ w_out  (4096 x 1024 x 1024).
  gemm_bf16<0, false, true, false><<<dim3(Ddim / GM_BN, M / GM_BM), 256, 0, stream>>>(
      act_b, wout_b, nullptr, hbuf, nullptr, (int)M, Ddim, Ddim);

  // 7) h = layernorm(a_proj + x); h f32 -> f32a, h bf16 -> act_b.
  residual_layernorm<<<(int)M, 256, 0, stream>>>(hbuf, x, ln1_w, ln1_b,
                                                 f32a, act_b);

  // 8) ffn1 = silu(h @ w1 + b1), bf16 only (4096 x 4096 x 1024).
  gemm_bf16<1, true, false, true><<<dim3(4 * Ddim / GM_BN, M / GM_BM), 256, 0, stream>>>(
      act_b, w1_b, b1, nullptr, ffn1_b, (int)M, 4 * Ddim, Ddim);

  // 9) ffn2 = ffn1 @ w2 + b2, f32 (4096 x 1024 x 4096).
  gemm_bf16<0, true, true, false><<<dim3(Ddim / GM_BN, M / GM_BM), 256, 0, stream>>>(
      ffn1_b, w2_b, b2, hbuf, nullptr, (int)M, Ddim, 4 * Ddim);

  // 10) out = layernorm(ffn2 + h).
  residual_layernorm<<<(int)M, 256, 0, stream>>>(hbuf, f32a, ln2_w, ln2_b,
                                                 out, nullptr);
}
